// CDVAE_7275674599805
// MI455X (gfx1250) — matrix operand order repaired and checked by reading, not compiled, not run on hardware
//
#include <hip/hip_runtime.h>
#include <hip/hip_bf16.h>
#include <cstdint>

#define NN 50000
#define NE 400000

typedef __attribute__((ext_vector_type(16))) _Float16 v16h;
typedef __attribute__((ext_vector_type(8)))  _Float16 v8h;
typedef __attribute__((ext_vector_type(8)))  float    v8f;

__device__ __forceinline__ float siluf(float x) { return x / (1.0f + __expf(-x)); }

__device__ __forceinline__ v8h cvt8g(const float* p) {
  const float4 x = *reinterpret_cast<const float4*>(p);
  const float4 y = *reinterpret_cast<const float4*>(p + 4);
  v8h r;
  r[0] = (_Float16)x.x; r[1] = (_Float16)x.y; r[2] = (_Float16)x.z; r[3] = (_Float16)x.w;
  r[4] = (_Float16)y.x; r[5] = (_Float16)y.y; r[6] = (_Float16)y.z; r[7] = (_Float16)y.w;
  return r;
}

__device__ __forceinline__ v16h mk16(v8h a, v8h b) {
  v16h r;
#pragma unroll
  for (int i = 0; i < 8; ++i) { r[i] = a[i]; r[i + 8] = b[i]; }
  return r;
}

__device__ __forceinline__ v8f wmma16(v16h a, v16h b, v8f c) {
  return __builtin_amdgcn_wmma_f32_16x16x32_f16(false, a, false, b, (short)0, c, false, false);
}

// column pointer into concat([node_feat[src], node_feat[dst], edge_attr]) for one edge
__device__ __forceinline__ const float* ecol(int c, const float* __restrict__ nf,
                                             const float* __restrict__ ea,
                                             int s, int d, int e) {
  if (c < 128)      return nf + (size_t)s * 128 + c;
  else if (c < 256) return nf + (size_t)d * 128 + (c - 128);
  else              return ea + (size_t)e * 64 + (c - 256);
}

// ---------------- setup kernels ----------------
__global__ void init_kernel(float* __restrict__ aggr, float* __restrict__ outC,
                            const float* __restrict__ coords, int nA, int nC) {
  int i = blockIdx.x * blockDim.x + threadIdx.x;
  if (i < nA) aggr[i] = 0.0f;
  if (i < nC) outC[i] = coords[i];
}

// W[K][N] (f32, row-major) -> WT[N][K] (f16)
__global__ void cvtWT_kernel(const float* __restrict__ W, _Float16* __restrict__ WT,
                             int K, int N, int total) {
  int i = blockIdx.x * blockDim.x + threadIdx.x;
  if (i >= total) return;
  int k = i / N, n = i % N;
  WT[(size_t)n * K + k] = (_Float16)W[i];
}

// ---------------- edge kernel: 4 waves/block, 16 edges/wave ----------------
__global__ __launch_bounds__(128) void egnn_edge(
    const float* __restrict__ nf, const float* __restrict__ ea,
    const int* __restrict__ ei, const float* __restrict__ coords,
    const _Float16* __restrict__ We1T, const _Float16* __restrict__ We2T,
    const _Float16* __restrict__ Wc1T,
    const float* __restrict__ be1, const float* __restrict__ be2,
    const float* __restrict__ bc1, const float* __restrict__ Wc2,
    const float* __restrict__ bc2,
    float* __restrict__ aggr, float* __restrict__ outC) {
  __shared__ __align__(32) _Float16 sH1[4][16][136];
  __shared__ __align__(32) _Float16 sMsg[4][16][136];
  __shared__ __align__(16) float    sH3[4][16][72];
  __shared__ int sDst[4][16];

  const int lane = threadIdx.x & 31;
  const int w    = threadIdx.x >> 5;
  const int hi   = lane >> 4;
  const int l15  = lane & 15;
  const int kbA  = hi ? 16 : 0;   // A-matrix K base (contiguous, matches B)
  const int kbB  = hi ? 16 : 0;   // B-matrix K base (32x16 f16 layout)

  const int eBase = (blockIdx.x * 4 + w) * 16;
  const int eMine = eBase + l15;                // row m = l15 for this lane
  const int s = ei[eMine];
  const int d = ei[NE + eMine];
  sDst[w][l15] = d;                             // both halves write same value
  __builtin_amdgcn_wave_barrier();

  // ---- layer 1: [16,320] x [320,128] ----
  v8f acc[8];
#pragma unroll
  for (int n = 0; n < 8; ++n) acc[n] = v8f{0, 0, 0, 0, 0, 0, 0, 0};
  for (int k = 0; k < 320; k += 32) {
    v8h a0 = cvt8g(ecol(k + kbA,      nf, ea, s, d, eMine));
    v8h a1 = cvt8g(ecol(k + kbA + 8,  nf, ea, s, d, eMine));
    v16h a = mk16(a0, a1);
#pragma unroll
    for (int n = 0; n < 8; ++n) {
      const v16h b = *reinterpret_cast<const v16h*>(We1T + (size_t)(n * 16 + l15) * 320 + k + kbB);
      acc[n] = wmma16(a, b, acc[n]);
    }
  }
#pragma unroll
  for (int n = 0; n < 8; ++n) {
    const int ng = n * 16 + l15;
    const float bb = be1[ng];
#pragma unroll
    for (int r = 0; r < 8; ++r)
      sH1[w][r + hi * 8][ng] = (_Float16)siluf(acc[n][r] + bb);
  }
  __builtin_amdgcn_wave_barrier();

  // ---- layer 2: [16,128] x [128,128] ----
  v8f acc2[8];
#pragma unroll
  for (int n = 0; n < 8; ++n) acc2[n] = v8f{0, 0, 0, 0, 0, 0, 0, 0};
  for (int k = 0; k < 128; k += 32) {
    v8h a0 = *reinterpret_cast<const v8h*>(&sH1[w][l15][k + kbA]);
    v8h a1 = *reinterpret_cast<const v8h*>(&sH1[w][l15][k + kbA + 8]);
    v16h a = mk16(a0, a1);
#pragma unroll
    for (int n = 0; n < 8; ++n) {
      const v16h b = *reinterpret_cast<const v16h*>(We2T + (size_t)(n * 16 + l15) * 128 + k + kbB);
      acc2[n] = wmma16(a, b, acc2[n]);
    }
  }
  // msg = silu(acc2 + be2): keep f32 in regs for scatter, stash f16 in LDS for coord MLP
#pragma unroll
  for (int n = 0; n < 8; ++n) {
    const int ng = n * 16 + l15;
    const float bb = be2[ng];
#pragma unroll
    for (int r = 0; r < 8; ++r) {
      float v = siluf(acc2[n][r] + bb);
      acc2[n][r] = v;
      sMsg[w][r + hi * 8][ng] = (_Float16)v;
    }
  }
  __builtin_amdgcn_wave_barrier();

  // ---- scatter messages: aggr[dst] += msg ----
#pragma unroll
  for (int n = 0; n < 8; ++n) {
    const int ng = n * 16 + l15;
#pragma unroll
    for (int r = 0; r < 8; ++r) {
      const int dn = sDst[w][r + hi * 8];
      atomicAdd(&aggr[(size_t)dn * 128 + ng], acc2[n][r]);
    }
  }

  // ---- coord MLP: [16,128] x [128,64], silu, then dot with Wc2 ----
  v8f acc3[4];
#pragma unroll
  for (int n = 0; n < 4; ++n) acc3[n] = v8f{0, 0, 0, 0, 0, 0, 0, 0};
  for (int k = 0; k < 128; k += 32) {
    v8h a0 = *reinterpret_cast<const v8h*>(&sMsg[w][l15][k + kbA]);
    v8h a1 = *reinterpret_cast<const v8h*>(&sMsg[w][l15][k + kbA + 8]);
    v16h a = mk16(a0, a1);
#pragma unroll
    for (int n = 0; n < 4; ++n) {
      const v16h b = *reinterpret_cast<const v16h*>(Wc1T + (size_t)(n * 16 + l15) * 128 + k + kbB);
      acc3[n] = wmma16(a, b, acc3[n]);
    }
  }
#pragma unroll
  for (int n = 0; n < 4; ++n) {
    const int ng = n * 16 + l15;
    const float bb = bc1[ng];
#pragma unroll
    for (int r = 0; r < 8; ++r)
      sH3[w][r + hi * 8][ng] = siluf(acc3[n][r] + bb);
  }
  __builtin_amdgcn_wave_barrier();

  if (lane < 16) {
    float cw = bc2[0];
#pragma unroll 8
    for (int j = 0; j < 64; ++j) cw += sH3[w][lane][j] * Wc2[j];
    const float dx = coords[3 * s + 0] - coords[3 * d + 0];
    const float dy = coords[3 * s + 1] - coords[3 * d + 1];
    const float dz = coords[3 * s + 2] - coords[3 * d + 2];
    const float inv = cw / (sqrtf(dx * dx + dy * dy + dz * dz) + 1e-8f);
    atomicAdd(&outC[3 * d + 0], inv * dx);
    atomicAdd(&outC[3 * d + 1], inv * dy);
    atomicAdd(&outC[3 * d + 2], inv * dz);
  }
}

// ---------------- node kernel: 4 waves/block, 16 nodes/wave ----------------
__global__ __launch_bounds__(128) void egnn_node(
    const float* __restrict__ nf, const float* __restrict__ aggr,
    const _Float16* __restrict__ Wn1T, const _Float16* __restrict__ Wn2T,
    const float* __restrict__ bn1, const float* __restrict__ bn2,
    float* __restrict__ outN) {
  __shared__ __align__(32) _Float16 sH[4][16][136];

  const int lane = threadIdx.x & 31;
  const int w    = threadIdx.x >> 5;
  const int hi   = lane >> 4;
  const int l15  = lane & 15;
  const int kbA  = hi ? 16 : 0;
  const int kbB  = hi ? 16 : 0;

  const int nBase = (blockIdx.x * 4 + w) * 16;
  int row = nBase + l15;
  const int rc = (row < NN) ? row : (NN - 1);   // clamp tail loads, keep EXEC full

  // ---- layer 1: [16,256] x [256,128]   (A = [node_feat | aggr]) ----
  v8f acc[8];
#pragma unroll
  for (int n = 0; n < 8; ++n) acc[n] = v8f{0, 0, 0, 0, 0, 0, 0, 0};
  for (int k = 0; k < 256; k += 32) {
    const int c0 = k + kbA, c1 = k + kbA + 8;
    const float* p0 = (c0 < 128) ? (nf + (size_t)rc * 128 + c0) : (aggr + (size_t)rc * 128 + (c0 - 128));
    const float* p1 = (c1 < 128) ? (nf + (size_t)rc * 128 + c1) : (aggr + (size_t)rc * 128 + (c1 - 128));
    v16h a = mk16(cvt8g(p0), cvt8g(p1));
#pragma unroll
    for (int n = 0; n < 8; ++n) {
      const v16h b = *reinterpret_cast<const v16h*>(Wn1T + (size_t)(n * 16 + l15) * 256 + k + kbB);
      acc[n] = wmma16(a, b, acc[n]);
    }
  }
#pragma unroll
  for (int n = 0; n < 8; ++n) {
    const int ng = n * 16 + l15;
    const float bb = bn1[ng];
#pragma unroll
    for (int r = 0; r < 8; ++r)
      sH[w][r + hi * 8][ng] = (_Float16)siluf(acc[n][r] + bb);
  }
  __builtin_amdgcn_wave_barrier();

  // ---- layer 2: [16,128] x [128,128] + residual ----
  v8f acc2[8];
#pragma unroll
  for (int n = 0; n < 8; ++n) acc2[n] = v8f{0, 0, 0, 0, 0, 0, 0, 0};
  for (int k = 0; k < 128; k += 32) {
    v8h a0 = *reinterpret_cast<const v8h*>(&sH[w][l15][k + kbA]);
    v8h a1 = *reinterpret_cast<const v8h*>(&sH[w][l15][k + kbA + 8]);
    v16h a = mk16(a0, a1);
#pragma unroll
    for (int n = 0; n < 8; ++n) {
      const v16h b = *reinterpret_cast<const v16h*>(Wn2T + (size_t)(n * 16 + l15) * 128 + k + kbB);
      acc2[n] = wmma16(a, b, acc2[n]);
    }
  }
#pragma unroll
  for (int n = 0; n < 8; ++n) {
    const int ng = n * 16 + l15;
    const float bb = bn2[ng];
#pragma unroll
    for (int r = 0; r < 8; ++r) {
      const int node = nBase + r + hi * 8;
      if (node < NN) {
        const size_t o = (size_t)node * 128 + ng;
        outN[o] = nf[o] + acc2[n][r] + bb;
      }
    }
  }
}

// ---------------- host launch ----------------
extern "C" void kernel_launch(void* const* d_in, const int* in_sizes, int n_in,
                              void* d_out, int out_size, void* d_ws, size_t ws_size,
                              hipStream_t stream) {
  const float* nf     = (const float*)d_in[0];
  const float* ea     = (const float*)d_in[1];
  const float* coords = (const float*)d_in[2];
  const int*   ei     = (const int*)d_in[3];
  const float* We1 = (const float*)d_in[4];
  const float* be1 = (const float*)d_in[5];
  const float* We2 = (const float*)d_in[6];
  const float* be2 = (const float*)d_in[7];
  const float* Wn1 = (const float*)d_in[8];
  const float* bn1 = (const float*)d_in[9];
  const float* Wn2 = (const float*)d_in[10];
  const float* bn2 = (const float*)d_in[11];
  const float* Wc1 = (const float*)d_in[12];
  const float* bc1 = (const float*)d_in[13];
  const float* Wc2 = (const float*)d_in[14];
  const float* bc2 = (const float*)d_in[15];

  float* outN = (float*)d_out;
  float* outC = outN + (size_t)NN * 128;

  // workspace layout (all offsets 32B-aligned)
  char* ws = (char*)d_ws;
  float*    aggr = (float*)ws;                          // 50000*128 f32 = 25,600,000 B
  _Float16* We1T = (_Float16*)(ws + 25600000);          // 128x320 f16 = 81,920 B
  _Float16* We2T = (_Float16*)(ws + 25600000 + 81920);  // 128x128 f16 = 32,768 B
  _Float16* Wn1T = (_Float16*)(ws + 25600000 + 81920 + 32768);            // 128x256 = 65,536 B
  _Float16* Wn2T = (_Float16*)(ws + 25600000 + 81920 + 32768 + 65536);    // 128x128 = 32,768 B
  _Float16* Wc1T = (_Float16*)(ws + 25600000 + 81920 + 32768 + 65536 + 32768); // 64x128 = 16,384 B

  const int nA = NN * 128;      // 6,400,000
  const int nC = NN * 3;        // 150,000
  init_kernel<<<(nA + 255) / 256, 256, 0, stream>>>(aggr, outC, coords, nA, nC);

  cvtWT_kernel<<<(40960 + 255) / 256, 256, 0, stream>>>(We1, We1T, 320, 128, 40960);
  cvtWT_kernel<<<(16384 + 255) / 256, 256, 0, stream>>>(We2, We2T, 128, 128, 16384);
  cvtWT_kernel<<<(32768 + 255) / 256, 256, 0, stream>>>(Wn1, Wn1T, 256, 128, 32768);
  cvtWT_kernel<<<(16384 + 255) / 256, 256, 0, stream>>>(Wn2, Wn2T, 128, 128, 16384);
  cvtWT_kernel<<<(8192 + 255) / 256, 256, 0, stream>>>(Wc1, Wc1T, 128, 64, 8192);

  egnn_edge<<<NE / 64, 128, 0, stream>>>(nf, ea, ei, coords, We1T, We2T, Wc1T,
                                         be1, be2, bc1, Wc2, bc2, aggr, outC);

  const int nodeBlocks = (NN + 63) / 64;  // 782
  egnn_node<<<nodeBlocks, 128, 0, stream>>>(nf, aggr, Wn1T, Wn2T, bn1, bn2, outN);
}
